// MultiHeadAttention_19396072309006
// MI455X (gfx1250) — matrix operand order repaired and checked by reading, not compiled, not run on hardware
//
#include <hip/hip_runtime.h>
#include <math.h>

// MI455X / gfx1250, wave32.
// GEMMs: TDM tensor_load_to_lds (double-buffered, HW row padding) -> v_wmma bf16.
static constexpr int Bb = 2, Ss = 2048, Ee = 1024, Hh = 16, Dk = 64;

typedef __attribute__((ext_vector_type(16))) __bf16 v16bf;
typedef __attribute__((ext_vector_type(8)))  float  v8f;
typedef __attribute__((ext_vector_type(4)))  unsigned su4;
typedef __attribute__((ext_vector_type(8)))  unsigned su8;

union FragB16 { v16bf v; uint4 q[2]; };
union Pack8   { uint4 q; unsigned short h[8]; };

__device__ __forceinline__ unsigned short f2bf(float f) {
  unsigned int u = __float_as_uint(f);
  unsigned int r = (u + 0x7FFFu + ((u >> 16) & 1u)) >> 16;  // RNE
  return (unsigned short)r;
}

__device__ __forceinline__ v8f wmma_bf16(const FragB16& a, const FragB16& b, v8f c) {
  return __builtin_amdgcn_wmma_f32_16x16x32_bf16(false, a.v, false, b.v,
                                                 (short)0, c, false, false);
}

// Low 32 bits of a flat LDS pointer are the LDS byte offset (ISA 10.2 aperture map).
__device__ __forceinline__ unsigned lds_addr(const void* p) {
  return (unsigned)(size_t)p;
}

// ---------------------------------------------------------------------------
// Tensor Data Mover: DMA one 128-row x 64-elem bf16 tile (row stride Ee) from
// global to LDS. HW pads +16B after every 128B row -> LDS row stride 144B,
// matching the 72-ushort padded layout the WMMA fragment readers use.
// D# group0/group1 per CDNA5 ISA. 2-D tile, so
// descriptor groups 2/3 are omitted (VADDR2/3 = NULL form).
// ---------------------------------------------------------------------------
__device__ __forceinline__ void tdm_tile_load(const unsigned short* g, unsigned ldsOff) {
  const unsigned long long ga = (unsigned long long)(size_t)g;
  su4 g0;
  g0[0] = 1u;                                   // count=1, user descriptor
  g0[1] = ldsOff;                               // lds_addr (bytes)
  g0[2] = (unsigned)ga;                         // global_addr[31:0]
  g0[3] = (unsigned)(ga >> 32) | (2u << 30);    // global_addr[56:32] | type=2
  su8 g1;
  g1[0] = (1u << 16)       // data_size = 2 bytes
        | (1u << 20)       // pad_enable
        | (4u << 22)       // pad_interval: every 32 DWORDs (= 128B row)
        | (3u << 25);      // pad_amount: 4 DWORDs (= 16B)
  const unsigned td0 = 1u << 20, td1 = 1u << 20;  // tensor dims (no OOB hit)
  g1[1] = (td0 & 0xFFFFu) << 16;                         // tensor_dim0 lo
  g1[2] = (td0 >> 16) | ((td1 & 0xFFFFu) << 16);         // dim0 hi | dim1 lo
  g1[3] = (td1 >> 16) | (64u << 16);                     // dim1 hi | tile_dim0=64
  g1[4] = 128u;                                          // tile_dim1=128, tile_dim2=0
  g1[5] = (unsigned)Ee;                                  // tensor_dim0_stride lo32
  g1[6] = 0u;                                            // stride0 hi | stride1 lo
  g1[7] = 0u;                                            // stride1 hi
  asm volatile("tensor_load_to_lds %0, %1" :: "s"(g0), "s"(g1) : "memory");
}

// ---------------------------------------------------------------------------
// f32 -> bf16 conversion pass (one-time; keeps the GEMM hot loop pure WMMA).
// ---------------------------------------------------------------------------
__global__ __launch_bounds__(256) void cvt_bf16(const float* __restrict__ in,
                                                unsigned short* __restrict__ out,
                                                int n8) {
  for (int i = blockIdx.x * blockDim.x + threadIdx.x; i < n8;
       i += gridDim.x * blockDim.x) {
    float4 f0 = ((const float4*)in)[i * 2];
    float4 f1 = ((const float4*)in)[i * 2 + 1];
    Pack8 pk;
    pk.h[0] = f2bf(f0.x); pk.h[1] = f2bf(f0.y);
    pk.h[2] = f2bf(f0.z); pk.h[3] = f2bf(f0.w);
    pk.h[4] = f2bf(f1.x); pk.h[5] = f2bf(f1.y);
    pk.h[6] = f2bf(f1.z); pk.h[7] = f2bf(f1.w);
    ((uint4*)out)[i] = pk.q;
  }
}

// ---------------------------------------------------------------------------
// GEMM: C[m,n] = scale * sum_k A[m,k] * W[n,k]   (y = A @ W.T), A/W bf16.
// Block tile 128x128, K-tile 64, 8 waves (2x4), wave tile 64x32.
// Tiles staged by the Tensor Data Mover, double-buffered; wave 0 issues the
// DMA and waits TENSORcnt, the workgroup barrier publishes the data.
// MODE 0: f32 row-major (final out) | 1: bf16 (B,H,S,DK) | 3: bf16 (B,H,DK,S)
// ---------------------------------------------------------------------------
template <int MODE>
__global__ __launch_bounds__(256) void gemm_wmma(const unsigned short* __restrict__ Ag,
                                                 const unsigned short* __restrict__ Wg,
                                                 void* __restrict__ Cg,
                                                 float scale) {
  constexpr int MT = 128, NT = 128, KT = 64, LDS_S = KT + 8;  // 72 bf16 = 144 B
  __shared__ __align__(16) unsigned short As[2][MT * LDS_S];
  __shared__ __align__(16) unsigned short Bs[2][NT * LDS_S];

  const int tid  = threadIdx.x;
  const int lane = tid & 31, w = tid >> 5;
  const int half = lane >> 4, lm = lane & 15;
  const int m0 = blockIdx.y * MT, n0 = blockIdx.x * NT;
  const int wm = (w >> 2) * 64, wn = (w & 3) * 32;

  v8f acc[4][2];
#pragma unroll
  for (int i = 0; i < 4; ++i)
#pragma unroll
    for (int j = 0; j < 2; ++j)
      acc[i][j] = (v8f){0.f, 0.f, 0.f, 0.f, 0.f, 0.f, 0.f, 0.f};

  int buf = 0;
  if (w == 0) {
    tdm_tile_load(&Ag[(size_t)m0 * Ee], lds_addr(As[0]));
    tdm_tile_load(&Wg[(size_t)n0 * Ee], lds_addr(Bs[0]));
    __builtin_amdgcn_s_wait_tensorcnt(0);
  }
  __syncthreads();

  for (int k0 = 0; k0 < Ee; k0 += KT) {
    if (w == 0 && k0 + KT < Ee) {  // DMA next k-tile into the other buffer
      tdm_tile_load(&Ag[(size_t)m0 * Ee + k0 + KT], lds_addr(As[buf ^ 1]));
      tdm_tile_load(&Wg[(size_t)n0 * Ee + k0 + KT], lds_addr(Bs[buf ^ 1]));
    }

    // Two 16x16x32 k-steps over the 64-wide tile.
#pragma unroll
    for (int s = 0; s < 2; ++s) {
      FragB16 a[4], b[2];
#pragma unroll
      for (int mi = 0; mi < 4; ++mi) {
        const int base = (wm + mi * 16 + lm) * LDS_S + s * 32;
        a[mi].q[0] = *(const uint4*)&As[buf][base + half * 8];
        a[mi].q[1] = *(const uint4*)&As[buf][base + 16 + half * 8];
      }
#pragma unroll
      for (int ni = 0; ni < 2; ++ni) {
        const int base = (wn + ni * 16 + lm) * LDS_S + s * 32;
        b[ni].q[0] = *(const uint4*)&Bs[buf][base + half * 8];
        b[ni].q[1] = *(const uint4*)&Bs[buf][base + 16 + half * 8];
      }
#pragma unroll
      for (int mi = 0; mi < 4; ++mi)
#pragma unroll
        for (int ni = 0; ni < 2; ++ni)
          acc[mi][ni] = wmma_bf16(a[mi], b[ni], acc[mi][ni]);
    }

    if (w == 0) __builtin_amdgcn_s_wait_tensorcnt(0);  // next tile landed
    __syncthreads();                                   // publish to all waves
    buf ^= 1;
  }

  // Epilogue. C layout: VGPR r -> M = r + half*8, lane%16 -> N.
#pragma unroll
  for (int mi = 0; mi < 4; ++mi)
#pragma unroll
    for (int ni = 0; ni < 2; ++ni)
#pragma unroll
      for (int r = 0; r < 8; ++r) {
        const int m = m0 + wm + mi * 16 + half * 8 + r;
        const int n = n0 + wn + ni * 16 + lm;
        const float val = acc[mi][ni][r] * scale;
        if (MODE == 0) {
          ((float*)Cg)[(size_t)m * Ee + n] = val;
        } else {
          const int bb = m >> 11, s = m & (Ss - 1);   // S = 2048
          const int h = n >> 6, dk = n & (Dk - 1);    // DK = 64
          size_t idx;
          if (MODE == 1) idx = (((size_t)(bb * Hh + h)) * Ss + s) * Dk + dk;
          else           idx = (((size_t)(bb * Hh + h)) * Dk + dk) * Ss + s;
          ((unsigned short*)Cg)[idx] = f2bf(val);
        }
      }
}

// ---------------------------------------------------------------------------
// Fused flash-attention: Q,K bf16 (B,H,S,DK), V bf16 (B,H,DK,S).
// Each wave owns 16 query rows; 8 independent waves per block (no barriers).
// Scores 16x32 per tile -> online softmax -> P via LDS transpose -> P@V.
// ---------------------------------------------------------------------------
__global__ __launch_bounds__(256) void attn_wmma(const unsigned short* __restrict__ Q,
                                                 const unsigned short* __restrict__ Kt,
                                                 const unsigned short* __restrict__ Vt,
                                                 unsigned short* __restrict__ Aout) {
  __shared__ __align__(16) unsigned short Ps[8][16 * 32];  // per-wave P tile

  const int tid  = threadIdx.x;
  const int lane = tid & 31, w = tid >> 5;
  const int half = lane >> 4, lm = lane & 15;
  const int bh = blockIdx.x >> 4;       // 16 q-tiles of 128 per (b,h)
  const int qt = blockIdx.x & 15;
  const int q0 = qt * 128 + w * 16;
  const int b = bh >> 4, h = bh & 15;   // H = 16
  const size_t base = (size_t)bh * Ss * Dk;

  FragB16 qa[2];
  {
    const unsigned short* qp = &Q[base + (size_t)(q0 + lm) * Dk];
#pragma unroll
    for (int f = 0; f < 2; ++f) {
      qa[f].q[0] = *(const uint4*)&qp[f * 32 + half * 8];
      qa[f].q[1] = *(const uint4*)&qp[f * 32 + 16 + half * 8];
    }
  }

  v8f o[4];
#pragma unroll
  for (int f = 0; f < 4; ++f) o[f] = (v8f){0.f, 0.f, 0.f, 0.f, 0.f, 0.f, 0.f, 0.f};
  float mr[8], lr[8];
#pragma unroll
  for (int r = 0; r < 8; ++r) { mr[r] = -INFINITY; lr[r] = 0.f; }

  for (int j0 = 0; j0 < Ss; j0 += 32) {
    // ---- scores S = Q K^T (pre-scaled in Q) ----
    v8f sc[2];
#pragma unroll
    for (int ni = 0; ni < 2; ++ni) {
      const unsigned short* kp = &Kt[base + (size_t)(j0 + ni * 16 + lm) * Dk];
      FragB16 kb0, kb1;
      kb0.q[0] = *(const uint4*)&kp[half * 8];
      kb0.q[1] = *(const uint4*)&kp[16 + half * 8];
      kb1.q[0] = *(const uint4*)&kp[32 + half * 8];
      kb1.q[1] = *(const uint4*)&kp[32 + 16 + half * 8];
      v8f z = (v8f){0.f, 0.f, 0.f, 0.f, 0.f, 0.f, 0.f, 0.f};
      z = wmma_bf16(qa[0], kb0, z);
      z = wmma_bf16(qa[1], kb1, z);
      sc[ni] = z;
    }

    // ---- online softmax: lane tracks rows r + half*8 ----
    float pm[8];
#pragma unroll
    for (int r = 0; r < 8; ++r) pm[r] = fmaxf(sc[0][r], sc[1][r]);
#pragma unroll
    for (int mask = 1; mask < 16; mask <<= 1)
#pragma unroll
      for (int r = 0; r < 8; ++r) pm[r] = fmaxf(pm[r], __shfl_xor(pm[r], mask, 32));
    float al[8];
#pragma unroll
    for (int r = 0; r < 8; ++r) {
      const float nm = fmaxf(mr[r], pm[r]);
      al[r] = __expf(mr[r] - nm);
      mr[r] = nm;
      lr[r] *= al[r];
    }
#pragma unroll
    for (int f = 0; f < 4; ++f)
#pragma unroll
      for (int r = 0; r < 8; ++r) o[f][r] *= al[r];

    float ps[8];
#pragma unroll
    for (int r = 0; r < 8; ++r) ps[r] = 0.f;
#pragma unroll
    for (int ni = 0; ni < 2; ++ni)
#pragma unroll
      for (int r = 0; r < 8; ++r) {
        const float p = __expf(sc[ni][r] - mr[r]);
        ps[r] += p;
        Ps[w][(half * 8 + r) * 32 + ni * 16 + lm] = f2bf(p);  // C -> row-major
      }
#pragma unroll
    for (int mask = 1; mask < 16; mask <<= 1)
#pragma unroll
      for (int r = 0; r < 8; ++r) ps[r] += __shfl_xor(ps[r], mask, 32);
#pragma unroll
    for (int r = 0; r < 8; ++r) lr[r] += ps[r];

    // ---- re-read P as a 16x32 A-fragment (per-wave LDS, in-order) ----
    FragB16 pa;
    pa.q[0] = *(const uint4*)&Ps[w][lm * 32 + half * 8];
    pa.q[1] = *(const uint4*)&Ps[w][lm * 32 + 16 + half * 8];

    // ---- O += P @ V : V^T rows are dk, contiguous keys ----
#pragma unroll
    for (int f = 0; f < 4; ++f) {
      const unsigned short* vp = &Vt[base + (size_t)(f * 16 + lm) * Ss + j0];
      FragB16 vb;
      vb.q[0] = *(const uint4*)&vp[half * 8];
      vb.q[1] = *(const uint4*)&vp[16 + half * 8];
      o[f] = wmma_bf16(pa, vb, o[f]);
    }
  }

#pragma unroll
  for (int f = 0; f < 4; ++f)
#pragma unroll
    for (int r = 0; r < 8; ++r) {
      const int s = q0 + half * 8 + r;
      const int e = h * 64 + f * 16 + lm;
      Aout[((size_t)(b * Ss + s)) * Ee + e] = f2bf(o[f][r] / lr[r]);
    }
}

// ---------------------------------------------------------------------------
extern "C" void kernel_launch(void* const* d_in, const int* in_sizes, int n_in,
                              void* d_out, int out_size, void* d_ws, size_t ws_size,
                              hipStream_t stream) {
  (void)in_sizes; (void)n_in; (void)out_size; (void)ws_size;
  const float* x  = (const float*)d_in[0];
  const float* Wq = (const float*)d_in[1];
  const float* Wk = (const float*)d_in[2];
  const float* Wv = (const float*)d_in[3];
  const float* Wo = (const float*)d_in[4];

  const size_t nX = (size_t)Bb * Ss * Ee;    // 4M
  const size_t nW = (size_t)Ee * Ee;         // 1M
  unsigned short* xb  = (unsigned short*)d_ws;
  unsigned short* Wqb = xb  + nX;
  unsigned short* Wkb = Wqb + nW;
  unsigned short* Wvb = Wkb + nW;
  unsigned short* Wob = Wvb + nW;
  unsigned short* Qb  = Wob + nW;
  unsigned short* Kb  = Qb + nX;
  unsigned short* Vb  = Kb + nX;
  unsigned short* At  = Vb + nX;             // total 48 MB of ws

  dim3 blk(256);
  // one-time f32 -> bf16 conversions
  cvt_bf16<<<dim3((int)(nX / 8 / 256)), blk, 0, stream>>>(x,  xb,  (int)(nX / 8));
  cvt_bf16<<<dim3((int)(nW / 8 / 256)), blk, 0, stream>>>(Wq, Wqb, (int)(nW / 8));
  cvt_bf16<<<dim3((int)(nW / 8 / 256)), blk, 0, stream>>>(Wk, Wkb, (int)(nW / 8));
  cvt_bf16<<<dim3((int)(nW / 8 / 256)), blk, 0, stream>>>(Wv, Wvb, (int)(nW / 8));
  cvt_bf16<<<dim3((int)(nW / 8 / 256)), blk, 0, stream>>>(Wo, Wob, (int)(nW / 8));

  dim3 grid(Ee / 128, (Bb * Ss) / 128);      // (8, 32)
  gemm_wmma<1><<<grid, blk, 0, stream>>>(xb, Wqb, Qb, 0.125f);  // 1/sqrt(64)
  gemm_wmma<1><<<grid, blk, 0, stream>>>(xb, Wkb, Kb, 1.0f);
  gemm_wmma<3><<<grid, blk, 0, stream>>>(xb, Wvb, Vb, 1.0f);    // V^T layout

  attn_wmma<<<dim3(Bb * Hh * (Ss / 128)), blk, 0, stream>>>(Qb, Kb, Vb, At);

  gemm_wmma<0><<<grid, blk, 0, stream>>>(At, Wob, d_out, 1.0f);
}
